// CrossAttentionLayer_6571299962915
// MI455X (gfx1250) — compile-verified, matrix operand order checked by reading, not run on hardware
//
#include <hip/hip_runtime.h>
#include <hip/hip_bf16.h>

// ---------------- common types / helpers ----------------
typedef __attribute__((ext_vector_type(16))) __bf16 v16bf;
typedef __attribute__((ext_vector_type(8)))  float  v8f;

union FragBF {
    uint4 q[2];
    v16bf v;
};

__device__ __forceinline__ unsigned short f2bf(float f) {
    union { float f; unsigned u; } v; v.f = f;
    unsigned u = v.u;
    u += 0x7FFFu + ((u >> 16) & 1u);   // round-to-nearest-even
    return (unsigned short)(u >> 16);
}
__device__ __forceinline__ unsigned pk2(float a, float b) {
    return (unsigned)f2bf(a) | ((unsigned)f2bf(b) << 16);
}

// LDS byte offset of a __shared__ object: generic LDS addresses carry the
// byte offset in addr[31:0] (ISA 10.2 aperture mapping).
__device__ __forceinline__ unsigned lds_off(const void* p) {
    return (unsigned)(size_t)p;
}

// async global -> LDS copy, 16 bytes per lane (ASYNCcnt-tracked)
__device__ __forceinline__ void async_b128(unsigned lds, const void* ga) {
    asm volatile("global_load_async_to_lds_b128 %0, %1, off"
                 :: "v"(lds), "v"(ga) : "memory");
}
__device__ __forceinline__ void wait_async0() {
    asm volatile("s_wait_asynccnt 0x0" ::: "memory");
}
__device__ __forceinline__ void wait_async6() {
    asm volatile("s_wait_asynccnt 0x6" ::: "memory");
}

// ---------------- bulk fp32 -> bf16 conversion (one-shot) ----------------
__global__ __launch_bounds__(256)
void cvt_bf16(const float* __restrict__ in, unsigned short* __restrict__ out)
{
    const size_t i = ((size_t)blockIdx.x * 256 + threadIdx.x) * 8;
    const float4 a = *(const float4*)(in + i);
    const float4 b = *(const float4*)(in + i + 4);
    uint4 o;
    o.x = pk2(a.x, a.y);
    o.y = pk2(a.z, a.w);
    o.z = pk2(b.x, b.y);
    o.w = pk2(b.z, b.w);
    *(uint4*)(out + i) = o;
}

// ---------------- GEMM: C[M,N] = X[M,K] * W[N,K]^T + bias (+ residual) ----------------
// X, W both bf16 row-major. 128x256 block tile, BK=32, 8 waves (2x4), 64x64 per wave.
// Double-buffered LDS with async global->LDS copies (ASYNCcnt pipelined).
template <bool OUTBF, bool RES>
__global__ __launch_bounds__(256)
void gemm_bf16(const unsigned short* __restrict__ X, const unsigned short* __restrict__ Wb,
               const float* __restrict__ bias, const float* __restrict__ Resid,
               void* __restrict__ Outv, int M, int N, int K)
{
    constexpr int LDA = 20;                    // uints per LDS row (16 used = 32 bf16)
    __shared__ unsigned lsA[2][128 * LDA];     // 2 x 10.2 KB
    __shared__ unsigned lsB[2][256 * LDA];     // 2 x 20.5 KB

    const int tid  = threadIdx.x;
    const int lane = tid & 31;
    const int w    = tid >> 5;
    const int wm   = w >> 2;                   // 0..1  (64-row slice)
    const int wn   = w & 3;                    // 0..3  (64-col slice)
    const int half = lane >> 4;
    const int ln15 = lane & 15;
    const int mBase = blockIdx.y * 128;
    const int nBase = blockIdx.x * 256;

    // staging thread mapping
    const int arow = tid >> 1;                 // 0..127, two threads per row
    const int aq   = (tid & 1) * 2;            // first uint4 slot of this thread
    const int brow = tid;                      // 0..255, one thread per row

    v8f acc[4][4] = {};

    auto issueA = [&](int k0, int buf) {
        unsigned lds = lds_off(&lsA[buf][arow * LDA + aq * 4]);
        const unsigned short* g = X + (size_t)(mBase + arow) * K + k0 + aq * 8;
        async_b128(lds,      g);
        async_b128(lds + 16, g + 8);
    };
    auto issueB = [&](int k0, int buf) {
        unsigned lds = lds_off(&lsB[buf][brow * LDA]);
        const unsigned short* g = Wb + (size_t)(nBase + brow) * K + k0;
        async_b128(lds,      g);
        async_b128(lds + 16, g + 8);
        async_b128(lds + 32, g + 16);
        async_b128(lds + 48, g + 24);
    };

    issueA(0, 0);
    issueB(0, 0);

    const int NT = K >> 5;
    for (int t = 0; t < NT; ++t) {
        const int cur = t & 1, nxt = cur ^ 1;
        if (t + 1 < NT) {
            issueA((t + 1) * 32, nxt);
            issueB((t + 1) * 32, nxt);
            wait_async6();                     // group t landed; group t+1 in flight
        } else {
            wait_async0();
        }
        __syncthreads();

        FragBF af[4], bfm[4];
        #pragma unroll
        for (int i = 0; i < 4; ++i) {
            int base = (wm * 64 + i * 16 + ln15) * LDA + half * 4;
            af[i].q[0] = *(const uint4*)&lsA[cur][base];
            af[i].q[1] = *(const uint4*)&lsA[cur][base + 8];
        }
        #pragma unroll
        for (int j = 0; j < 4; ++j) {
            int base = (wn * 64 + j * 16 + ln15) * LDA + half * 4;
            bfm[j].q[0] = *(const uint4*)&lsB[cur][base];
            bfm[j].q[1] = *(const uint4*)&lsB[cur][base + 8];
        }
        #pragma unroll
        for (int i = 0; i < 4; ++i)
            #pragma unroll
            for (int j = 0; j < 4; ++j)
                acc[i][j] = __builtin_amdgcn_wmma_f32_16x16x32_bf16(
                    false, af[i].v, false, bfm[j].v, (short)0, acc[i][j], false, false);

        __syncthreads();                       // all waves done reading 'cur'
    }

    // ---- epilogue ----
    #pragma unroll
    for (int i = 0; i < 4; ++i) {
        #pragma unroll
        for (int j = 0; j < 4; ++j) {
            int   col = nBase + wn * 64 + j * 16 + ln15;
            float bb  = bias[col];
            #pragma unroll
            for (int g = 0; g < 8; ++g) {
                int   rowg = mBase + wm * 64 + i * 16 + g + half * 8;
                float v    = acc[i][j][g] + bb;
                if (RES) v += Resid[(size_t)rowg * N + col];
                if (OUTBF) ((unsigned short*)Outv)[(size_t)rowg * N + col] = f2bf(v);
                else       ((float*)Outv)[(size_t)rowg * N + col] = v;
            }
        }
    }
}

// ---------------- flash-style cross attention ----------------
// grid (L/64, NH, B), 128 threads (4 waves, 16 Q-rows each). HD = 128 fixed.
__global__ __launch_bounds__(128)
void attn_kernel(const unsigned short* __restrict__ Q,
                 const unsigned short* __restrict__ Kb,
                 const unsigned short* __restrict__ Vb,
                 const int* __restrict__ mask,
                 unsigned short* __restrict__ ctx,
                 int B, int L, int V, int H)
{
    const int b     = blockIdx.z;
    const int h     = blockIdx.y;
    const int qTile = blockIdx.x * 64;
    const int tid   = threadIdx.x;
    const int lane  = tid & 31;
    const int w     = tid >> 5;
    const int half  = lane >> 4;
    const int ln15  = lane & 15;

    constexpr int LQ = 68;   // uints/row for 128-bf16 rows (64 used)
    constexpr int LV = 36;   // uints/row for 64-bf16 rows (32 used)
    __shared__ unsigned lsQ [64 * LQ];
    __shared__ unsigned lsK [64 * LQ];
    __shared__ unsigned lsVt[128 * LV];      // V transposed: [d][v]
    __shared__ unsigned lsP [4 * 16 * LV];   // per-wave softmax probs (bf16)

    // ---- async load Q tile (64 x 128 bf16): half a row (128B) per thread ----
    {
        int row = tid >> 1;
        int qh  = (tid & 1) * 8;
        unsigned ldsq = lds_off(&lsQ[row * LQ + qh * 4]);
        const unsigned short* g = Q + (size_t)(b * L + qTile + row) * H + h * 128 + qh * 8;
        #pragma unroll
        for (int i = 0; i < 8; ++i)
            async_b128(ldsq + i * 16, g + i * 8);
    }
    wait_async0();
    __syncthreads();

    FragBF qf[4];
    #pragma unroll
    for (int kk = 0; kk < 4; ++kk) {
        int base = (w * 16 + ln15) * LQ + kk * 16 + half * 4;
        qf[kk].q[0] = *(const uint4*)&lsQ[base];
        qf[kk].q[1] = *(const uint4*)&lsQ[base + 8];
    }

    float m_run[8], l_run[8];
    v8f   of[8] = {};
    #pragma unroll
    for (int g = 0; g < 8; ++g) { m_run[g] = -1e30f; l_run[g] = 0.f; }

    for (int vb = 0; vb < V; vb += 64) {
        __syncthreads();   // prior iteration done with lsK / lsVt
        // ---- K tile (64 x 128 bf16) via async copy ----
        {
            int row = tid >> 1;
            int qh  = (tid & 1) * 8;
            unsigned ldsk = lds_off(&lsK[row * LQ + qh * 4]);
            const unsigned short* g = Kb + (size_t)(b * V + vb + row) * H + h * 128 + qh * 8;
            #pragma unroll
            for (int i = 0; i < 8; ++i)
                async_b128(ldsk + i * 16, g + i * 8);
        }
        // ---- V tile transposed into lsVt[d][v] (manual: transpose needed) ----
        {
            int v  = tid >> 1;
            int dh = (tid & 1) * 64;
            const uint4* src = (const uint4*)(Vb + (size_t)(b * V + vb + v) * H + h * 128 + dh);
            unsigned short* vt = (unsigned short*)lsVt;
            #pragma unroll
            for (int i = 0; i < 8; ++i) {
                uint4 q4 = src[i];
                unsigned uu[4] = {q4.x, q4.y, q4.z, q4.w};
                #pragma unroll
                for (int c = 0; c < 4; ++c) {
                    int d0 = dh + i * 8 + c * 2;
                    vt[(size_t)d0 * (LV * 2) + v]       = (unsigned short)(uu[c] & 0xFFFFu);
                    vt[(size_t)(d0 + 1) * (LV * 2) + v] = (unsigned short)(uu[c] >> 16);
                }
            }
        }
        wait_async0();
        __syncthreads();

        // ---- S = Q K^T (16 x 64 per wave) ----
        v8f s[4] = {};
        #pragma unroll
        for (int kk = 0; kk < 4; ++kk) {
            #pragma unroll
            for (int j = 0; j < 4; ++j) {
                FragBF kf;
                int base = (j * 16 + ln15) * LQ + kk * 16 + half * 4;
                kf.q[0] = *(const uint4*)&lsK[base];
                kf.q[1] = *(const uint4*)&lsK[base + 8];
                s[j] = __builtin_amdgcn_wmma_f32_16x16x32_bf16(
                    false, qf[kk].v, false, kf.v, (short)0, s[j], false, false);
            }
        }
        // ---- scale + mask ----
        const float sc = 0.088388347648318447f;   // 1/sqrt(128)
        int mk[4];
        #pragma unroll
        for (int j = 0; j < 4; ++j) mk[j] = mask[(size_t)b * V + vb + j * 16 + ln15];
        #pragma unroll
        for (int j = 0; j < 4; ++j)
            #pragma unroll
            for (int g = 0; g < 8; ++g) {
                float x = s[j][g] * sc;
                s[j][g] = (mk[j] != 0) ? x : -1e30f;
            }
        // ---- online softmax ----
        float corr[8], rs[8];
        #pragma unroll
        for (int g = 0; g < 8; ++g) {
            float x = fmaxf(fmaxf(s[0][g], s[1][g]), fmaxf(s[2][g], s[3][g]));
            #pragma unroll
            for (int off = 1; off < 16; off <<= 1)
                x = fmaxf(x, __shfl_xor(x, off, 32));
            float mn  = fmaxf(m_run[g], x);
            corr[g]   = __expf(m_run[g] - mn);
            m_run[g]  = mn;
        }
        #pragma unroll
        for (int g = 0; g < 8; ++g) {
            float sum = 0.f;
            #pragma unroll
            for (int j = 0; j < 4; ++j) {
                float p = __expf(s[j][g] - m_run[g]);
                s[j][g] = p;
                sum += p;
            }
            #pragma unroll
            for (int off = 1; off < 16; off <<= 1)
                sum += __shfl_xor(sum, off, 32);
            rs[g] = sum;
        }
        #pragma unroll
        for (int g = 0; g < 8; ++g) l_run[g] = l_run[g] * corr[g] + rs[g];
        #pragma unroll
        for (int f = 0; f < 8; ++f)
            #pragma unroll
            for (int g = 0; g < 8; ++g) of[f][g] *= corr[g];

        // ---- P -> LDS as bf16 (per-wave region), then reload as A-fragments ----
        {
            unsigned short* pp = (unsigned short*)&lsP[w * 16 * LV];
            #pragma unroll
            for (int j = 0; j < 4; ++j)
                #pragma unroll
                for (int g = 0; g < 8; ++g) {
                    int row = g + half * 8;
                    int col = j * 16 + ln15;
                    pp[row * (LV * 2) + col] = f2bf(s[j][g]);
                }
        }
        asm volatile("s_wait_dscnt 0" ::: "memory");

        // ---- O += P * V ----
        #pragma unroll
        for (int kk2 = 0; kk2 < 2; ++kk2) {
            FragBF pa;
            {
                int base = (w * 16 + ln15) * LV + kk2 * 16 + half * 4;
                pa.q[0] = *(const uint4*)&lsP[base];
                pa.q[1] = *(const uint4*)&lsP[base + 8];
            }
            #pragma unroll
            for (int n = 0; n < 8; ++n) {
                FragBF vfr;
                int base = (n * 16 + ln15) * LV + kk2 * 16 + half * 4;
                vfr.q[0] = *(const uint4*)&lsVt[base];
                vfr.q[1] = *(const uint4*)&lsVt[base + 8];
                of[n] = __builtin_amdgcn_wmma_f32_16x16x32_bf16(
                    false, pa.v, false, vfr.v, (short)0, of[n], false, false);
            }
        }
    }

    // ---- epilogue: normalize and store ctx (bf16) ----
    #pragma unroll
    for (int n = 0; n < 8; ++n)
        #pragma unroll
        for (int g = 0; g < 8; ++g) {
            int   l   = qTile + w * 16 + g + half * 8;
            int   d   = n * 16 + ln15;
            float val = of[n][g] / l_run[g];
            ctx[(size_t)(b * L + l) * H + h * 128 + d] = f2bf(val);
        }
}

// ---------------- layernorm (residual already fused upstream) ----------------
__global__ __launch_bounds__(256)
void ln_kernel(const float* __restrict__ X, const float* __restrict__ gamma,
               const float* __restrict__ beta, float* __restrict__ out, int H)
{
    __shared__ float red[16];
    const int row  = blockIdx.x;
    const float* x = X + (size_t)row * H;
    float sum = 0.f, sq = 0.f;
    for (int i = threadIdx.x * 4; i < H; i += 1024) {
        float4 v = *(const float4*)(x + i);
        sum += v.x + v.y + v.z + v.w;
        sq  += v.x * v.x + v.y * v.y + v.z * v.z + v.w * v.w;
    }
    for (int off = 1; off < 32; off <<= 1) {
        sum += __shfl_xor(sum, off, 32);
        sq  += __shfl_xor(sq,  off, 32);
    }
    const int w = threadIdx.x >> 5, lane = threadIdx.x & 31;
    if (lane == 0) { red[w] = sum; red[8 + w] = sq; }
    __syncthreads();
    if (w == 0) {
        float s = (lane < 8) ? red[lane]     : 0.f;
        float q = (lane < 8) ? red[8 + lane] : 0.f;
        for (int off = 1; off < 8; off <<= 1) {
            s += __shfl_xor(s, off, 32);
            q += __shfl_xor(q, off, 32);
        }
        if (lane == 0) { red[0] = s; red[1] = q; }
    }
    __syncthreads();
    const float mu  = red[0] / H;
    const float var = red[1] / H - mu * mu;
    const float inv = rsqrtf(var + 1e-5f);
    for (int i = threadIdx.x * 4; i < H; i += 1024) {
        float4 v = *(const float4*)(x + i);
        float4 o;
        o.x = (v.x - mu) * inv * gamma[i]     + beta[i];
        o.y = (v.y - mu) * inv * gamma[i + 1] + beta[i + 1];
        o.z = (v.z - mu) * inv * gamma[i + 2] + beta[i + 2];
        o.w = (v.w - mu) * inv * gamma[i + 3] + beta[i + 3];
        *(float4*)(out + (size_t)row * H + i) = o;
    }
}

// ---------------- launcher ----------------
extern "C" void kernel_launch(void* const* d_in, const int* in_sizes, int n_in,
                              void* d_out, int out_size, void* d_ws, size_t ws_size,
                              hipStream_t stream)
{
    (void)in_sizes; (void)n_in; (void)out_size; (void)ws_size;
    const int B = 2, L = 1024, V = 1024, H = 4096;
    const int M = B * L;                       // 2048 (also B*V)
    const size_t NEL = (size_t)M * H;          // 8,388,608
    const size_t NW  = (size_t)H * H;          // 16,777,216

    const float* hs  = (const float*)d_in[0];
    const float* vfe = (const float*)d_in[1];
    const int*   msk = (const int*)  d_in[2];
    const float* Wq  = (const float*)d_in[3];
    const float* bq  = (const float*)d_in[4];
    const float* Wk  = (const float*)d_in[5];
    const float* bk  = (const float*)d_in[6];
    const float* Wv  = (const float*)d_in[7];
    const float* bv  = (const float*)d_in[8];
    const float* Wo  = (const float*)d_in[9];
    const float* bo  = (const float*)d_in[10];
    const float* lng = (const float*)d_in[11];
    const float* lnb = (const float*)d_in[12];

    unsigned short* p   = (unsigned short*)d_ws;
    unsigned short* hsb = p; p += NEL;
    unsigned short* vfb = p; p += NEL;
    unsigned short* Wqb = p; p += NW;
    unsigned short* Wkb = p; p += NW;
    unsigned short* Wvb = p; p += NW;
    unsigned short* Wob = p; p += NW;
    unsigned short* Qb  = p; p += NEL;
    unsigned short* Kb  = p; p += NEL;
    unsigned short* Vb  = p; p += NEL;
    unsigned short* Cx  = p; p += NEL;
    float*          Xr  = (float*)p;

    // one-shot bf16 conversions (2048 elements per block)
    cvt_bf16<<<(unsigned)(NEL / 2048), 256, 0, stream>>>(hs,  hsb);
    cvt_bf16<<<(unsigned)(NEL / 2048), 256, 0, stream>>>(vfe, vfb);
    cvt_bf16<<<(unsigned)(NW  / 2048), 256, 0, stream>>>(Wq,  Wqb);
    cvt_bf16<<<(unsigned)(NW  / 2048), 256, 0, stream>>>(Wk,  Wkb);
    cvt_bf16<<<(unsigned)(NW  / 2048), 256, 0, stream>>>(Wv,  Wvb);
    cvt_bf16<<<(unsigned)(NW  / 2048), 256, 0, stream>>>(Wo,  Wob);

    dim3 gg(H / 256, M / 128);                 // (16, 16)
    gemm_bf16<true,  false><<<gg, 256, 0, stream>>>(hsb, Wqb, bq, nullptr, Qb, M, H, H);
    gemm_bf16<true,  false><<<gg, 256, 0, stream>>>(vfb, Wkb, bk, nullptr, Kb, M, H, H);
    gemm_bf16<true,  false><<<gg, 256, 0, stream>>>(vfb, Wvb, bv, nullptr, Vb, M, H, H);

    attn_kernel<<<dim3(L / 64, 32, B), 128, 0, stream>>>(Qb, Kb, Vb, msk, Cx, B, L, V, H);

    gemm_bf16<false, true ><<<gg, 256, 0, stream>>>(Cx, Wob, bo, hs, Xr, M, H, H);

    ln_kernel<<<M, 256, 0, stream>>>(Xr, lng, lnb, (float*)d_out, H);
}
